// CrossCorrV2_58686433132755
// MI455X (gfx1250) — matrix operand order repaired and checked by reading, not compiled, not run on hardware
//
#include <hip/hip_runtime.h>
#include <hip/hip_bf16.h>

// ---------------------------------------------------------------------------
// CrossCorrV2 for MI455X (gfx1250, wave32, WMMA)
//
// Reference collapses: top_k(k=100) over 100 patches + mean == mean over all
// patches; conv and bilinear resize are linear => out = resize(corr(x1n, Kavg)/36).
// We compute per-pixel tap responses U[pos, t] = sum_c x1n[pos,c]*K[t,c] as a
// dense f16 WMMA GEMM (M=16384, N=48 (36 taps + 12 zero pads), K=64), then a
// shifted 36-tap gather builds the 129x129 conv output, then bilinear 129->128.
// ---------------------------------------------------------------------------

typedef __attribute__((ext_vector_type(16))) _Float16 v16h;
typedef __attribute__((ext_vector_type(8)))  _Float16 v8h;
typedef __attribute__((ext_vector_type(8)))  float    v8f;

#define B_      8
#define C_      64
#define H_      128
#define W_      128
#define HW_     (H_ * W_)           // 16384
#define H2_     60
#define HW2_    (H2_ * H2_)         // 3600
#define NTAP    36
#define NTAPP   48                  // padded to 3 WMMA N-tiles
#define HO_     129
#define HWO_    (HO_ * HO_)         // 16641

// workspace layout (bytes), all 256-aligned
#define KF_OFF    0u                                  // f16 [B][48][64]
#define KF_BYTES  (B_ * NTAPP * C_ * 2u)              // 49152
#define X1N_OFF   (KF_OFF + KF_BYTES)                 // f16 [B][16384][64]
#define X1N_BYTES (B_ * (unsigned)HW_ * C_ * 2u)      // 16777216
#define U_OFF     (X1N_OFF + X1N_BYTES)               // f32 [B][16384][48]
#define U_BYTES   (B_ * (unsigned)HW_ * NTAPP * 4u)   // 25165824
#define O129_OFF  (U_OFF + U_BYTES)                   // f32 [B][129][129]

__device__ __forceinline__ int iclamp(int v, int lo, int hi) {
    return v < lo ? lo : (v > hi ? hi : v);
}

// ---------------------------------------------------------------------------
// K1: per-batch averaged, normalized filter.
// rn2[pos] = 1/||x2[b,:,pos]|| (in LDS), then
// K[b,t=(i,j),c] = (1/(100*36)) * sum_{p0,p1} x2[b,c,6p0+i,6p1+j]*rn2[...]
// (the 1/36 from the reference's /(hh*ww) is folded in). Pads t=36..47 -> 0.
// ---------------------------------------------------------------------------
__global__ void cc_prep_filter(const float* __restrict__ x2,
                               _Float16* __restrict__ kf) {
    __shared__ float rn2[HW2_];
    const int b = blockIdx.x;
    const int tid = threadIdx.x;
    const float* xb = x2 + (size_t)b * C_ * HW2_;

    for (int p = tid; p < HW2_; p += blockDim.x) {
        float s = 0.f;
        #pragma unroll 8
        for (int c = 0; c < C_; ++c) {
            float v = xb[c * HW2_ + p];
            s += v * v;
        }
        rn2[p] = rsqrtf(s);
    }
    __syncthreads();

    for (int idx = tid; idx < NTAPP * C_; idx += blockDim.x) {
        const int t = idx >> 6;
        const int c = idx & 63;
        float s = 0.f;
        if (t < NTAP) {
            const int i = t / 6, j = t % 6;
            const float* xc = xb + c * HW2_;
            for (int p0 = 0; p0 < 10; ++p0) {
                const int row = (6 * p0 + i) * H2_;
                #pragma unroll
                for (int p1 = 0; p1 < 10; ++p1) {
                    const int pos = row + 6 * p1 + j;
                    s += xc[pos] * rn2[pos];
                }
            }
            s *= (1.0f / (100.0f * 36.0f));
        }
        kf[((size_t)b * NTAPP + t) * C_ + c] = (_Float16)s;
    }
}

// ---------------------------------------------------------------------------
// K2: channel-wise L2 normalize x1, store f16 channels-last [b][pos][c].
// ---------------------------------------------------------------------------
__global__ void cc_norm_x1(const float* __restrict__ x1,
                           _Float16* __restrict__ x1n) {
    const int gid = blockIdx.x * blockDim.x + threadIdx.x;
    if (gid >= B_ * HW_) return;
    const int b = gid >> 14;
    const int p = gid & (HW_ - 1);
    const float* xb = x1 + (size_t)b * C_ * HW_ + p;

    float v[C_];
    float s = 0.f;
    #pragma unroll 8
    for (int c = 0; c < C_; ++c) {
        v[c] = xb[c * HW_];
        s += v[c] * v[c];
    }
    const float r = rsqrtf(s);

    _Float16* o = x1n + ((size_t)gid << 6);
    #pragma unroll
    for (int c8 = 0; c8 < 8; ++c8) {
        v8h h;
        #pragma unroll
        for (int k = 0; k < 8; ++k)
            h[k] = (_Float16)(v[c8 * 8 + k] * r);
        *(v8h*)(o + c8 * 8) = h;
    }
}

// ---------------------------------------------------------------------------
// K3: WMMA tap-GEMM.  U[b][pos][t] = sum_c x1n[b][pos][c] * kf[b][t][c]
// One wave per (b, 16-row M tile); 3 N-tiles x 2 K-steps = 6 v_wmma per wave.
// Per-lane VGPR layouts (wave32): contiguous k = 16h + s in both A and B.
// ---------------------------------------------------------------------------
struct APair { v8h lo; v8h hi; };

__global__ void cc_wmma_tapgemm(const _Float16* __restrict__ x1n,
                                const _Float16* __restrict__ kf,
                                float* __restrict__ U) {
    const int tid   = threadIdx.x;
    const int lane  = tid & 31;
    const int wave  = blockIdx.x * (blockDim.x >> 5) + (tid >> 5); // 0..8191
    const int b     = wave >> 10;
    const int mtile = wave & 1023;
    const int l15   = lane & 15;
    const bool hi   = lane >= 16;

    const _Float16* aBase = x1n + ((size_t)(b * HW_ + mtile * 16 + l15) << 6);
    const _Float16* kb    = kf + (size_t)b * NTAPP * C_;
    __builtin_prefetch(kb, 0, 3);

    v8f acc0 = {}, acc1 = {}, acc2 = {};

    #pragma unroll
    for (int ks = 0; ks < 2; ++ks) {
        const int k0 = ks * 32;
        // A 16x32 f16: lanes 0-15 hold K {k0+0..15}; lanes 16-31
        // hold K {k0+16..31}.  Two aligned 16B loads per lane.
        APair ap;
        ap.lo = *(const v8h*)(aBase + k0 + (hi ? 16 : 0));
        ap.hi = *(const v8h*)(aBase + k0 + 8 + (hi ? 16 : 0));
        const v16h A = __builtin_bit_cast(v16h, ap);

        // B 32x16 f16: lane n (0-15) holds col N=n, rows K=k0+0..15;
        // lanes 16-31 hold col N=n-16, rows K=k0+16..31.  32B contiguous.
        const int krow = k0 + (hi ? 16 : 0);
        const v16h B0 = *(const v16h*)(kb + (size_t)( 0 + l15) * C_ + krow);
        const v16h B1 = *(const v16h*)(kb + (size_t)(16 + l15) * C_ + krow);
        const v16h B2 = *(const v16h*)(kb + (size_t)(32 + l15) * C_ + krow);

        acc0 = __builtin_amdgcn_wmma_f32_16x16x32_f16(false, A, false, B0,
                                                      (short)0, acc0, false, false);
        acc1 = __builtin_amdgcn_wmma_f32_16x16x32_f16(false, A, false, B1,
                                                      (short)0, acc1, false, false);
        acc2 = __builtin_amdgcn_wmma_f32_16x16x32_f16(false, A, false, B2,
                                                      (short)0, acc2, false, false);
    }

    // D layout: VGPR i -> M = i (lanes 0-15) / 8+i (lanes 16-31), N = lane&15.
    const int rowBase = b * HW_ + mtile * 16;
    #pragma unroll
    for (int i = 0; i < 8; ++i) {
        const int m = hi ? 8 + i : i;
        float* urow = U + (size_t)(rowBase + m) * NTAPP + l15;
        urow[0]  = acc0[i];
        urow[16] = acc1[i];
        urow[32] = acc2[i];
    }
}

// ---------------------------------------------------------------------------
// K4: conv output from tap responses:
// out129[b,y,x] = sum_{i,j} U[b, (y+i-3)*128 + (x+j-3), i*6+j]  (zero-padded)
// ---------------------------------------------------------------------------
__global__ void cc_gather129(const float* __restrict__ U,
                             float* __restrict__ out129) {
    const int gid = blockIdx.x * blockDim.x + threadIdx.x;
    if (gid >= B_ * HWO_) return;
    const int b = gid / HWO_;
    const int rem = gid - b * HWO_;
    const int y = rem / HO_;
    const int x = rem - y * HO_;

    const float* Ub = U + (size_t)b * HW_ * NTAPP;
    float s = 0.f;
    #pragma unroll
    for (int i = 0; i < 6; ++i) {
        const int yy = y + i - 3;
        if (yy < 0 || yy >= H_) continue;
        #pragma unroll
        for (int j = 0; j < 6; ++j) {
            const int xx = x + j - 3;
            if (xx < 0 || xx >= W_) continue;
            s += Ub[(size_t)(yy * W_ + xx) * NTAPP + (i * 6 + j)];
        }
    }
    out129[gid] = s;
}

// ---------------------------------------------------------------------------
// K5: half-pixel bilinear resize 129x129 -> 128x128 (align_corners=False).
// src = (dst + 0.5) * 129/128 - 0.5
// ---------------------------------------------------------------------------
__global__ void cc_resize_out(const float* __restrict__ out129,
                              float* __restrict__ out) {
    const int gid = blockIdx.x * blockDim.x + threadIdx.x;
    if (gid >= B_ * HW_) return;
    const int b = gid >> 14;
    const int rem = gid & (HW_ - 1);
    const int y = rem >> 7;
    const int x = rem & 127;

    const float sc  = 129.0f / 128.0f;
    const float off = 0.5f * sc - 0.5f; // 0.00390625
    const float fy = y * sc + off;
    const float fx = x * sc + off;
    int y0 = (int)floorf(fy);
    int x0 = (int)floorf(fx);
    const float wy = fy - (float)y0;
    const float wx = fx - (float)x0;
    const int y1 = iclamp(y0 + 1, 0, HO_ - 1);
    const int x1 = iclamp(x0 + 1, 0, HO_ - 1);
    y0 = iclamp(y0, 0, HO_ - 1);
    x0 = iclamp(x0, 0, HO_ - 1);

    const float* src = out129 + (size_t)b * HWO_;
    const float v00 = src[y0 * HO_ + x0];
    const float v01 = src[y0 * HO_ + x1];
    const float v10 = src[y1 * HO_ + x0];
    const float v11 = src[y1 * HO_ + x1];
    const float top = v00 + wx * (v01 - v00);
    const float bot = v10 + wx * (v11 - v10);
    out[gid] = top + wy * (bot - top);
}

// ---------------------------------------------------------------------------
extern "C" void kernel_launch(void* const* d_in, const int* in_sizes, int n_in,
                              void* d_out, int out_size, void* d_ws, size_t ws_size,
                              hipStream_t stream) {
    (void)in_sizes; (void)n_in; (void)out_size; (void)ws_size;
    const float* x1 = (const float*)d_in[0]; // [8,64,128,128]
    const float* x2 = (const float*)d_in[1]; // [8,64,60,60]
    float* out = (float*)d_out;              // [8,1,128,128]

    char* ws = (char*)d_ws;
    _Float16* kf     = (_Float16*)(ws + KF_OFF);
    _Float16* x1n    = (_Float16*)(ws + X1N_OFF);
    float*    U      = (float*)   (ws + U_OFF);
    float*    out129 = (float*)   (ws + O129_OFF);

    // K1: averaged normalized filter per batch
    cc_prep_filter<<<B_, 256, 0, stream>>>(x2, kf);

    // K2: normalize x1 -> f16 channels-last
    cc_norm_x1<<<(B_ * HW_ + 255) / 256, 256, 0, stream>>>(x1, x1n);

    // K3: WMMA tap GEMM (8 waves/block, 1 M-tile per wave, 8192 waves)
    cc_wmma_tapgemm<<<(B_ * (HW_ / 16)) / 8, 256, 0, stream>>>(x1n, kf, U);

    // K4: shifted 36-tap gather -> 129x129 conv output
    cc_gather129<<<(B_ * HWO_ + 255) / 256, 256, 0, stream>>>(U, out129);

    // K5: bilinear 129 -> 128
    cc_resize_out<<<(B_ * HW_ + 255) / 256, 256, 0, stream>>>(out129, out);
}
